// InitLayer_70866960384259
// MI455X (gfx1250) — matrix operand order repaired and checked by reading, not compiled, not run on hardware
//
#include <hip/hip_runtime.h>
#include <hip/hip_bf16.h>
#include <math.h>

typedef __attribute__((ext_vector_type(16))) _Float16 v16h;
typedef __attribute__((ext_vector_type(8)))  float    v8f;

#define E_EDGES 640000
#define N_NODES 20000
#define TILES   40000
#define NWAVES  6
#define BLK_EDGE (NWAVES * 32)
#define GRID_EDGE 1024

// d_out element offsets
#define OFF_NODE  81920000ull   // E*128
#define OFF_EDGEF 87680000ull   // + N*288
#define OFF_CUT   272000000ull  // + E*288
#define OFF_ACT   272640000ull  // + E

// ---------------- WMMA helpers ----------------

__device__ __forceinline__ v8f wmma_f16(v16h a, v16h b, v8f c) {
  return __builtin_amdgcn_wmma_f32_16x16x32_f16(false, a, false, b, (short)0, c,
                                                false, false);
}

// B fragment: 32x16 f16, lane = column (lane&15), K-half by lane>>4,
// 16 contiguous f16 (32B, 16B aligned) -> 2x ds_load_b128
__device__ __forceinline__ v16h load_b(const _Float16* p) {
  v16h r;
#pragma unroll
  for (int i = 0; i < 16; ++i) r[i] = p[i];
  return r;
}
__device__ __forceinline__ v16h load_b_masked(const _Float16* p, bool on) {
  v16h r;
#pragma unroll
  for (int i = 0; i < 16; ++i) r[i] = on ? p[i] : (_Float16)0.0f;
  return r;
}

// A fragment: 16x32 f16, lane = row (lane&15); elements 0..15 -> K=kh*16+0..15
// (one 32B run)
__device__ __forceinline__ v16h load_a(const _Float16* p) {
  v16h r;
#pragma unroll
  for (int i = 0; i < 16; ++i) r[i] = p[i];
  return r;
}

// hi/lo split GEMM over one 16-edge tile:  C[NT] += A(16 x 32*KT) * W
// acts: row-major stride 160 (per-wave LDS). weights: col-major stride KSTR.
template <int KT, int NT, int KSTR, bool PART>
__device__ __forceinline__ void gemm_tile(const _Float16* aH, const _Float16* aL,
                                          const _Float16* wH, const _Float16* wL,
                                          int lane, v8f* c) {
  const int ln = lane & 15, kh = lane >> 4;
#pragma unroll
  for (int kt = 0; kt < KT; ++kt) {
    const int aoff = ln * 160 + kt * 32 + kh * 16;
    v16h ah = load_a(aH + aoff);
    v16h al = load_a(aL + aoff);
#pragma unroll
    for (int nt = 0; nt < NT; ++nt) {
      const int boff = (nt * 16 + ln) * KSTR + kt * 32 + kh * 16;
      v16h bh, bl;
      if (PART && kt == KT - 1) {           // layer-1 tail: K=128..135 real only
        bool on = (kh == 0);                // lanes>=16 would read K>=144
        bh = load_b_masked(wH + boff, on);
        bl = load_b_masked(wL + boff, on);
      } else {
        bh = load_b(wH + boff);
        bl = load_b(wL + boff);
      }
      c[nt] = wmma_f16(ah, bh, c[nt]);
      c[nt] = wmma_f16(ah, bl, c[nt]);
      c[nt] = wmma_f16(al, bh, c[nt]);
    }
  }
}

__device__ __forceinline__ void split_store(float v, _Float16* hp, _Float16* lp) {
  _Float16 h = (_Float16)v;
  *hp = h;
  *lp = (_Float16)(v - (float)h);
}

template <int NT>
__device__ __forceinline__ void silu_store(v8f* c, _Float16* aH, _Float16* aL,
                                           int lane) {
  const int ln = lane & 15, kh = lane >> 4;
#pragma unroll
  for (int nt = 0; nt < NT; ++nt)
#pragma unroll
    for (int i = 0; i < 8; ++i) {
      float x = c[nt][i];
      float y = 1.6791770458221436f * x / (1.0f + __expf(-x));
      int row = i + kh * 8, col = nt * 16 + ln;
      split_store(y, &aH[row * 160 + col], &aL[row * 160 + col]);
    }
}

// ---------------- Edge pipeline kernel ----------------

__global__ __launch_bounds__(BLK_EDGE) void edge_kernel(
    const float* __restrict__ one_hot, const float* __restrict__ elen,
    const float* __restrict__ esh, const float* __restrict__ bfreq,
    const float* __restrict__ W1, const float* __restrict__ W2,
    const float* __restrict__ W3, const float* __restrict__ We,
    float* __restrict__ out_lat, float* __restrict__ out_edgef,
    float* __restrict__ out_cut, int* __restrict__ out_act) {
  __shared__ __align__(32) _Float16 smem[163328];  // 326,656 B
  _Float16* w1h = smem;              _Float16* w1l = smem + 18432;   // 128 x 144
  _Float16* w2h = smem + 36864;      _Float16* w2l = smem + 54272;   // 128 x 136
  _Float16* w3h = smem + 71680;      _Float16* w3l = smem + 89088;   // 128 x 136
  _Float16* weh = smem + 106496;     _Float16* wel = smem + 119552;  //  96 x 136
  const int tid = threadIdx.x, lane = tid & 31, wid = tid >> 5;
  _Float16* aH = smem + 132608 + wid * 5120;  // 16 x 160 acts (hi)
  _Float16* aL = aH + 2560;                   // 16 x 160 acts (lo)
  float* wbuf = (float*)aH;                   // aliases acts after W_env GEMM

  // ---- convert weights to hi/lo f16, column-major, fold 1/sqrt(K) ----
  const float s1 = 0.08574929257125441f;  // 1/sqrt(136)
  const float s2 = 0.08838834764831845f;  // 1/sqrt(128)
  for (int idx = tid; idx < 128 * 144; idx += BLK_EDGE) {
    int n = idx / 144, k = idx - n * 144;
    float v = (k < 136) ? W1[k * 128 + n] * s1 : 0.0f;
    split_store(v, &w1h[idx], &w1l[idx]);
  }
  for (int idx = tid; idx < 128 * 136; idx += BLK_EDGE) {
    int n = idx / 136, k = idx - n * 136;
    float v2 = (k < 128) ? W2[k * 128 + n] * s2 : 0.0f;
    float v3 = (k < 128) ? W3[k * 128 + n] * s2 : 0.0f;
    split_store(v2, &w2h[idx], &w2l[idx]);
    split_store(v3, &w3h[idx], &w3l[idx]);
  }
  for (int idx = tid; idx < 96 * 136; idx += BLK_EDGE) {
    int n = idx / 136, k = idx - n * 136;
    float v = (k < 128) ? We[k * 96 + n] * s2 : 0.0f;
    split_store(v, &weh[idx], &wel[idx]);
  }
  __syncthreads();

  const int ln = lane & 15, kh = lane >> 4;

  for (int tile = blockIdx.x * NWAVES + wid; tile < TILES;
       tile += gridDim.x * NWAVES) {
    const int e0 = tile * 16;

    // ---- build feat = [one_hot(128) | bessel(8) | 0(24)] in acts ----
#pragma unroll 1
    for (int row = 0; row < 16; ++row) {
      const float4 v =
          *(const float4*)(one_hot + (size_t)(e0 + row) * 128 + lane * 4);
      int b = row * 160 + lane * 4;
      split_store(v.x, &aH[b + 0], &aL[b + 0]);
      split_store(v.y, &aH[b + 1], &aL[b + 1]);
      split_store(v.z, &aH[b + 2], &aL[b + 2]);
      split_store(v.w, &aH[b + 3], &aL[b + 3]);
    }
    float mycut = 0.0f;
    if (lane < 16) {
      float r = elen[e0 + lane];
#pragma unroll
      for (int b = 0; b < 8; ++b) {
        float v = 0.6324555320336759f * sinf(bfreq[b] * r * 0.2f) / r;
        split_store(v, &aH[lane * 160 + 128 + b], &aL[lane * 160 + 128 + b]);
      }
      float x = r * 0.2f;
      float x2 = x * x, x4 = x2 * x2, x6 = x4 * x2;
      float cut = 1.0f - 28.0f * x6 + 48.0f * x6 * x - 21.0f * x6 * x2;
      mycut = (x < 1.0f) ? cut : 0.0f;
      out_cut[e0 + lane] = mycut;
      out_act[e0 + lane] = e0 + lane;
    } else {
      int m = lane - 16;
#pragma unroll
      for (int k = 136; k < 160; ++k) {
        aH[m * 160 + k] = (_Float16)0.0f;
        aL[m * 160 + k] = (_Float16)0.0f;
      }
    }

    v8f c[8];
    // ---- layer 1 : (16x136) @ (136x128), K padded to 160 ----
#pragma unroll
    for (int nt = 0; nt < 8; ++nt)
#pragma unroll
      for (int i = 0; i < 8; ++i) c[nt][i] = 0.0f;
    gemm_tile<5, 8, 144, true>(aH, aL, w1h, w1l, lane, c);
    silu_store<8>(c, aH, aL, lane);

    // ---- layer 2 ----
#pragma unroll
    for (int nt = 0; nt < 8; ++nt)
#pragma unroll
      for (int i = 0; i < 8; ++i) c[nt][i] = 0.0f;
    gemm_tile<4, 8, 136, false>(aH, aL, w2h, w2l, lane, c);
    silu_store<8>(c, aH, aL, lane);

    // ---- layer 3 -> latents = cutoff * new_lat ----
#pragma unroll
    for (int nt = 0; nt < 8; ++nt)
#pragma unroll
      for (int i = 0; i < 8; ++i) c[nt][i] = 0.0f;
    gemm_tile<4, 8, 136, false>(aH, aL, w3h, w3l, lane, c);

    float cutr[8];
#pragma unroll
    for (int i = 0; i < 8; ++i) cutr[i] = __shfl(mycut, kh * 8 + i, 32);
#pragma unroll
    for (int nt = 0; nt < 8; ++nt)
#pragma unroll
      for (int i = 0; i < 8; ++i) {
        int row = i + kh * 8, col = nt * 16 + ln;
        float v = c[nt][i] * cutr[i];
        out_lat[(size_t)(e0 + row) * 128 + col] = v;
        split_store(v, &aH[row * 160 + col], &aL[row * 160 + col]);
      }

    // ---- W_env : (16x128) @ (128x96) ----
    v8f d[6];
#pragma unroll
    for (int nt = 0; nt < 6; ++nt)
#pragma unroll
      for (int i = 0; i < 8; ++i) d[nt][i] = 0.0f;
    gemm_tile<4, 6, 136, false>(aH, aL, weh, wel, lane, d);

#pragma unroll
    for (int nt = 0; nt < 6; ++nt)
#pragma unroll
      for (int i = 0; i < 8; ++i) {
        int row = i + kh * 8, col = nt * 16 + ln;
        wbuf[row * 96 + col] = d[nt][i];  // acts dead; alias is safe
      }

    // ---- expand with spherical harmonics -> edge_features (288) ----
#pragma unroll 1
    for (int row = 0; row < 16; ++row) {
      const size_t e = (size_t)(e0 + row);
      const float* shp = esh + e * 9;
      const float* wr = wbuf + row * 96;
      float* outp = out_edgef + e * 288;
      outp[lane] = wr[lane] * shp[0];                       // f0 (32)
#pragma unroll
      for (int q = 1; q < 4; ++q) {                         // f1 (96)
        int t = lane + (q - 1) * 32;
        int m = t / 3, cc = t - 3 * m;
        outp[32 + t] = wr[32 + m] * shp[1 + cc];
      }
#pragma unroll
      for (int q = 0; q < 5; ++q) {                         // f2 (160)
        int t = lane + q * 32;
        int m = t / 5, cc = t - 5 * m;
        outp[128 + t] = wr[64 + m] * shp[4 + cc];
      }
    }
  }
}

// ---------------- segment-sum support (counting sort) ----------------

__global__ void hist_kernel(const int* __restrict__ center,
                            int* __restrict__ counts) {
  int e = blockIdx.x * 256 + threadIdx.x;
  if (e < E_EDGES) atomicAdd(&counts[center[e]], 1);
}

__global__ void scan_kernel(const int* __restrict__ counts,
                            int* __restrict__ offs, int n) {
  __shared__ int buf[1024];
  __shared__ int carry;
  int tid = threadIdx.x;
  if (tid == 0) carry = 0;
  __syncthreads();
  for (int base = 0; base < n; base += 1024) {
    int i = base + tid;
    int v = (i < n) ? counts[i] : 0;
    buf[tid] = v;
    __syncthreads();
    for (int off = 1; off < 1024; off <<= 1) {
      int t = (tid >= off) ? buf[tid - off] : 0;
      __syncthreads();
      buf[tid] += t;
      __syncthreads();
    }
    int incl = buf[tid];
    if (i < n) offs[i] = carry + incl - v;
    __syncthreads();
    if (tid == 1023) carry += incl;
    __syncthreads();
  }
  if (tid == 0) offs[n] = carry;
}

__global__ void scatter_kernel(const int* __restrict__ center,
                               const int* __restrict__ offs,
                               int* __restrict__ cursor,
                               int* __restrict__ elist) {
  int e = blockIdx.x * 256 + threadIdx.x;
  if (e < E_EDGES) {
    int cc = center[e];
    int pos = atomicAdd(&cursor[cc], 1);
    elist[offs[cc] + pos] = e;
  }
}

// ---------------- node gather + norms (one wave per node) ----------------

__global__ __launch_bounds__(256) void node_kernel(
    const float* __restrict__ edgef, const int* __restrict__ offs,
    const int* __restrict__ elist, const float* __restrict__ gamma_s,
    const float* __restrict__ beta_s, const float* __restrict__ gamma_v,
    float* __restrict__ node_out) {
  const int lane = threadIdx.x & 31, wid = threadIdx.x >> 5;
  const int node = blockIdx.x * 8 + wid;
  if (node >= N_NODES) return;
  float acc[9];
#pragma unroll
  for (int q = 0; q < 9; ++q) acc[q] = 0.0f;
  const int beg = offs[node], end = offs[node + 1];
  for (int i = beg; i < end; ++i) {
    const float* p = edgef + (size_t)elist[i] * 288;
#pragma unroll
    for (int q = 0; q < 9; ++q) acc[q] += p[lane + q * 32];
  }
  const float sc = 0.22360679774997896f;  // 1/sqrt(20)
#pragma unroll
  for (int q = 0; q < 9; ++q) acc[q] *= sc;

  // scalar layernorm over 32 channels
  float s = acc[0];
  float sum = s;
#pragma unroll
  for (int m = 1; m < 32; m <<= 1) sum += __shfl_xor(sum, m, 32);
  s -= sum * 0.03125f;
  float sq = s * s;
#pragma unroll
  for (int m = 1; m < 32; m <<= 1) sq += __shfl_xor(sq, m, 32);
  s = s / sqrtf(sq * 0.03125f + 1e-8f);
  s = s * gamma_s[lane] + beta_s[lane];
  node_out[(size_t)node * 288 + lane] = s;

  // vector norms
  float ss1 = acc[1] * acc[1] + acc[2] * acc[2] + acc[3] * acc[3];
  float ss2 = acc[4] * acc[4] + acc[5] * acc[5] + acc[6] * acc[6] +
              acc[7] * acc[7] + acc[8] * acc[8];
#pragma unroll
  for (int m = 1; m < 32; m <<= 1) {
    ss1 += __shfl_xor(ss1, m, 32);
    ss2 += __shfl_xor(ss2, m, 32);
  }
  float var1 = ss1 / 96.0f, var2 = ss2 / 160.0f;
  float inv = 1.0f / sqrtf(0.5f * (var1 + var2) + 1e-8f);
#pragma unroll
  for (int q = 1; q < 4; ++q) {
    int t = lane + (q - 1) * 32;
    int m = t / 3;
    node_out[(size_t)node * 288 + 32 + t] = acc[q] * inv * gamma_v[m];
  }
#pragma unroll
  for (int q = 4; q < 9; ++q) {
    int t = lane + (q - 4) * 32;
    int m = t / 5;
    node_out[(size_t)node * 288 + 128 + t] = acc[q] * inv * gamma_v[32 + m];
  }
}

// ---------------- host launcher ----------------

extern "C" void kernel_launch(void* const* d_in, const int* in_sizes, int n_in,
                              void* d_out, int out_size, void* d_ws,
                              size_t ws_size, hipStream_t stream) {
  const int* eidx = (const int*)d_in[0];       // (2,E): row0 = centers
  const float* esh = (const float*)d_in[3];    // (E,9)
  const float* elen = (const float*)d_in[4];   // (E,)
  const float* ehot = (const float*)d_in[5];   // (E,128)
  const float* W1 = (const float*)d_in[6];
  const float* W2 = (const float*)d_in[7];
  const float* W3 = (const float*)d_in[8];
  const float* We = (const float*)d_in[9];
  const float* bfreq = (const float*)d_in[10];
  const float* gs = (const float*)d_in[11];
  const float* bs = (const float*)d_in[12];
  const float* gv = (const float*)d_in[13];

  float* out = (float*)d_out;
  int* ws = (int*)d_ws;
  int* counts = ws;                 // N
  int* cursor = ws + N_NODES;       // N
  int* offs = ws + 2 * N_NODES;     // N+1
  int* elist = ws + 3 * N_NODES + 1;  // E

  hipMemsetAsync(ws, 0, 2 * N_NODES * sizeof(int), stream);
  hist_kernel<<<2500, 256, 0, stream>>>(eidx, counts);
  scan_kernel<<<1, 1024, 0, stream>>>(counts, offs, N_NODES);
  scatter_kernel<<<2500, 256, 0, stream>>>(eidx, offs, cursor, elist);
  edge_kernel<<<GRID_EDGE, BLK_EDGE, 0, stream>>>(
      ehot, elen, esh, bfreq, W1, W2, W3, We, out /*latents*/,
      out + OFF_EDGEF, out + OFF_CUT, (int*)(out + OFF_ACT));
  node_kernel<<<2500, 256, 0, stream>>>(out + OFF_EDGEF, offs, elist, gs, bs,
                                        gv, out + OFF_NODE);
}
